// SpikingDSABlock_68401649156428
// MI455X (gfx1250) — compile-verified, matrix operand order checked by reading, not run on hardware
//
#include <hip/hip_runtime.h>
#include <hip/hip_bf16.h>

typedef unsigned short u16;
typedef unsigned int   u32;
typedef unsigned long long u64;
typedef __attribute__((ext_vector_type(16))) __bf16 v16bf;
typedef __attribute__((ext_vector_type(8)))  float  v8f;
typedef __attribute__((ext_vector_type(4)))  unsigned int v4u;
typedef __attribute__((ext_vector_type(8)))  int v8i;
typedef __attribute__((ext_vector_type(4)))  int v4i;

#define B_  4
#define S_  1024
#define D_  1024
#define H_  16
#define DH_ 64
#define F_  4096

__device__ __forceinline__ u16 f2bf(float f) {
  u32 u = __float_as_uint(f);
  u32 r = u + 0x7FFFu + ((u >> 16) & 1u);   // round-to-nearest-even
  return (u16)(r >> 16);
}

// ---- TDM: DMA one [rows x 32] bf16 tile (row stride lda elems) into LDS ----
// LDS layout gets 16B padding after each 64B row => row stride 40 elems (LKP).
// D# per CDNA5 ISA ch.8: group0 = {count, lds_addr, global_addr, type=2},
// group1 = {data_size=2B, pad_enable, pad_interval(64B), pad_amount(4 dwords),
//           tensor_dim0/1, tile_dim0=32, tile_dim1=rows, dim0_stride=lda}.
// This toolchain exposes the 6-arg builtin: (g0, g1, g2, g3, g_extra, cpol).
__device__ __forceinline__ void tdm_load_tile(const u16* gptr, u32 lds_off,
                                              int rows, int lda_elems) {
  u64 ga = (u64)(const void*)gptr;
  v4u g0;
  g0[0] = 1u;                                        // count=1, user mode
  g0[1] = lds_off;                                   // lds_addr [63:32]
  g0[2] = (u32)(ga & 0xFFFFFFFFu);                   // global_addr lo
  g0[3] = (u32)((ga >> 32) & 0x01FFFFFFu) | (2u << 30);  // addr hi + type=2
  const u32 dim0 = 32u;                 // tensor sized exactly to the tile
  const u32 dim1 = (u32)rows;
  const u64 st0  = (u64)lda_elems;      // dim0 stride (data_size units)
  v8i g1;
  g1[0] = (int)((1u << 16) |            // data_size = 2 bytes
                (1u << 20) |            // pad_enable
                (3u << 22) |            // pad_interval: 2^3 * 8B = 64B
                (3u << 25));            // pad_amount: 4 DWORDs = 16B
  g1[1] = (int)((dim0 & 0xFFFFu) << 16);                       // dim0[15:0]
  g1[2] = (int)(((dim0 >> 16) & 0xFFFFu) | ((dim1 & 0xFFFFu) << 16));
  g1[3] = (int)(((dim1 >> 16) & 0xFFFFu) | (32u << 16));       // tile_dim0=32
  g1[4] = (int)(dim1 & 0xFFFFu);                               // tile_dim1=rows
  g1[5] = (int)(st0 & 0xFFFFFFFFu);
  g1[6] = (int)((st0 >> 32) & 0xFFFFu);                        // dim1_stride=0
  g1[7] = 0;
  v4i z4 = {0, 0, 0, 0};
  v8i z8 = {0, 0, 0, 0, 0, 0, 0, 0};
  __builtin_amdgcn_tensor_load_to_lds(g0, g1, z4, z4, z8, 0);
}

// ---------------- fp32 -> bf16 elementwise ----------------
__global__ __launch_bounds__(256) void cvt_bf16_kernel(const float* __restrict__ in,
                                                       u16* __restrict__ out, int n) {
  int i = blockIdx.x * 256 + threadIdx.x;
  if (i < n) out[i] = f2bf(in[i]);
}

// ---------------- fp32 [K,N] -> bf16 transposed [N,K] ----------------
__global__ __launch_bounds__(256) void cvtT_bf16_kernel(const float* __restrict__ in,
                                                        u16* __restrict__ out, int K, int N) {
  long long i = (long long)blockIdx.x * 256 + threadIdx.x;
  if (i < (long long)K * N) {
    int k = (int)(i / N), n = (int)(i % N);
    out[(long long)n * K + k] = f2bf(in[i]);
  }
}

// ---------------- WMMA bf16 GEMM: C = A[M,K] * B + bias (+res) ----------------
// B given as Bt row-major [N,K] (B_ROWMAJOR=false, TDM+double-buffer path) or
// B row-major [K,N] (true, transpose-in-LDS staging path).
template<int BM, int BN, int WAVES_M, int WAVES_N,
         bool B_ROWMAJOR, bool ADD_BIAS, bool RES_ADD, bool OUT_BF16, bool CAUSAL>
__global__ __launch_bounds__(256) void gemm_bf16_wmma(
    const u16* __restrict__ A, long long sAz, int lda,
    const u16* __restrict__ Bm, long long sBz, int ldb,
    const float* __restrict__ bias,
    const float* __restrict__ res, long long sRz, int ldr,
    void* __restrict__ Cout, long long sCz, int ldc,
    int K)
{
  constexpr int LKP  = 40;                        // padded LDS K-stride
  constexpr int NBUF = B_ROWMAJOR ? 1 : 2;        // double-buffer on TDM path
  constexpr int WTM  = BM / (WAVES_M * 16);
  constexpr int WTN  = BN / (WAVES_N * 16);
  __shared__ u16 lA[NBUF * BM * LKP];
  __shared__ u16 lB[NBUF * BN * LKP];

  const long long rowBase = (long long)blockIdx.y * BM;
  const long long colBase = (long long)blockIdx.x * BN;
  if (CAUSAL && colBase > rowBase + (BM - 1)) return;   // masked-out tile

  A  += (long long)blockIdx.z * sAz;
  Bm += (long long)blockIdx.z * sBz;
  const int tid  = threadIdx.x;
  const int wave = tid >> 5, lane = tid & 31;
  const int wm = wave / WAVES_N, wn = wave % WAVES_N;
  const int hl = lane >> 4, l16 = lane & 15;

  v8f acc[WTM][WTN] = {};
  union FragU { uint4 q[2]; v16bf v; };

  if constexpr (!B_ROWMAJOR) {
    // ---- TDM double-buffered pipeline ----
    const bool issuer = (wave == 0);
    const int nsteps = K >> 5;
    if (issuer) {
      tdm_load_tile(&A[rowBase * lda],  (u32)(uintptr_t)&lA[0], BM, lda);
      tdm_load_tile(&Bm[colBase * ldb], (u32)(uintptr_t)&lB[0], BN, ldb);
    }
    for (int s = 0; s < nsteps; ++s) {
      const int cur = s & 1;
      if (issuer) __builtin_amdgcn_s_wait_tensorcnt(0);
      __syncthreads();                               // tile ready, prior reads done
      if (issuer && (s + 1) < nsteps) {              // prefetch next tile via DMA
        tdm_load_tile(&A[rowBase * lda + (s + 1) * 32],
                      (u32)(uintptr_t)&lA[(cur ^ 1) * BM * LKP], BM, lda);
        tdm_load_tile(&Bm[colBase * ldb + (s + 1) * 32],
                      (u32)(uintptr_t)&lB[(cur ^ 1) * BN * LKP], BN, ldb);
      }
      const int bA = cur * BM * LKP, bB = cur * BN * LKP;
      FragU fa[WTM], fb[WTN];
#pragma unroll
      for (int i = 0; i < WTM; ++i) {
        int m = (wm * WTM + i) * 16 + l16;
        fa[i].q[0] = *(const uint4*)&lA[bA + m * LKP + hl * 8];
        fa[i].q[1] = *(const uint4*)&lA[bA + m * LKP + 16 + hl * 8];
      }
#pragma unroll
      for (int j = 0; j < WTN; ++j) {
        int n = (wn * WTN + j) * 16 + l16;
        fb[j].q[0] = *(const uint4*)&lB[bB + n * LKP + hl * 8];
        fb[j].q[1] = *(const uint4*)&lB[bB + n * LKP + 16 + hl * 8];
      }
#pragma unroll
      for (int i = 0; i < WTM; ++i)
#pragma unroll
        for (int j = 0; j < WTN; ++j)
          acc[i][j] = __builtin_amdgcn_wmma_f32_16x16x32_bf16(
              false, fa[i].v, false, fb[j].v, (short)0, acc[i][j], false, false);
      // no trailing barrier: next iteration's wait+barrier fences buffer reuse
    }
  } else {
    // ---- synchronous path with B transpose in LDS (P @ V) ----
    for (int k0 = 0; k0 < K; k0 += 32) {
      for (int c = tid; c < BM * 4; c += 256) {
        int r = c >> 2, kc = (c & 3) << 3;
        *(uint4*)&lA[r * LKP + kc] = *(const uint4*)&A[(rowBase + r) * lda + k0 + kc];
      }
      for (int e = tid; e < BN * 32; e += 256) {
        int k = e / BN, n = e % BN;
        lB[n * LKP + k] = Bm[(long long)(k0 + k) * ldb + colBase + n];
      }
      __syncthreads();
      FragU fa[WTM], fb[WTN];
#pragma unroll
      for (int i = 0; i < WTM; ++i) {
        int m = (wm * WTM + i) * 16 + l16;
        fa[i].q[0] = *(const uint4*)&lA[m * LKP + hl * 8];
        fa[i].q[1] = *(const uint4*)&lA[m * LKP + 16 + hl * 8];
      }
#pragma unroll
      for (int j = 0; j < WTN; ++j) {
        int n = (wn * WTN + j) * 16 + l16;
        fb[j].q[0] = *(const uint4*)&lB[n * LKP + hl * 8];
        fb[j].q[1] = *(const uint4*)&lB[n * LKP + 16 + hl * 8];
      }
#pragma unroll
      for (int i = 0; i < WTM; ++i)
#pragma unroll
        for (int j = 0; j < WTN; ++j)
          acc[i][j] = __builtin_amdgcn_wmma_f32_16x16x32_bf16(
              false, fa[i].v, false, fb[j].v, (short)0, acc[i][j], false, false);
      __syncthreads();
    }
  }

  // epilogue: C layout row = r + 8*half, col = lane%16
#pragma unroll
  for (int i = 0; i < WTM; ++i) {
#pragma unroll
    for (int j = 0; j < WTN; ++j) {
      long long col = colBase + (wn * WTN + j) * 16 + l16;
      float bv = ADD_BIAS ? bias[col] : 0.0f;
#pragma unroll
      for (int r = 0; r < 8; ++r) {
        long long row = rowBase + (wm * WTM + i) * 16 + hl * 8 + r;
        float v = acc[i][j][r] + bv;
        if (RES_ADD) v += res[(long long)blockIdx.z * sRz + row * ldr + col];
        long long co = (long long)blockIdx.z * sCz + row * ldc + col;
        if (OUT_BF16) ((u16*)Cout)[co] = f2bf(v);
        else          ((float*)Cout)[co] = v;
      }
    }
  }
}

// ---------------- causal softmax: fp32 scores -> bf16 probs ----------------
__global__ __launch_bounds__(256) void softmax_causal_kernel(const float* __restrict__ sc,
                                                             u16* __restrict__ p) {
  const int q = blockIdx.x, h = blockIdx.y;
  const float* row = sc + ((long long)h * S_ + q) * S_;
  u16* orow = p + ((long long)h * S_ + q) * S_;
  const int len = q + 1;
  const int tid = threadIdx.x;
  __shared__ float red[256];
  float mx = -1e30f;
  for (int i = tid; i < len; i += 256) mx = fmaxf(mx, row[i] * 0.125f);
  red[tid] = mx; __syncthreads();
  for (int s = 128; s > 0; s >>= 1) { if (tid < s) red[tid] = fmaxf(red[tid], red[tid + s]); __syncthreads(); }
  mx = red[0]; __syncthreads();
  float sum = 0.f;
  for (int i = tid; i < len; i += 256) sum += __expf(row[i] * 0.125f - mx);
  red[tid] = sum; __syncthreads();
  for (int s = 128; s > 0; s >>= 1) { if (tid < s) red[tid] += red[tid + s]; __syncthreads(); }
  const float inv = 1.0f / red[0];
  for (int i = tid; i < S_; i += 256) {
    float v = (i < len) ? __expf(row[i] * 0.125f - mx) * inv : 0.0f;
    orow[i] = f2bf(v);
  }
}

// ---------------- layernorm over D=1024, optional bf16 copy ----------------
template<bool OUTB>
__global__ __launch_bounds__(256) void layernorm_kernel(const float* __restrict__ X,
                                                        const float* __restrict__ g,
                                                        const float* __restrict__ bb,
                                                        float* __restrict__ Y,
                                                        u16* __restrict__ Yb) {
  const long long row = blockIdx.x;
  const float* x = X + row * D_;
  const int tid = threadIdx.x;
  float4 v4 = *(const float4*)&x[tid * 4];
  float vv[4] = { v4.x, v4.y, v4.z, v4.w };
  float s = vv[0] + vv[1] + vv[2] + vv[3];
  float s2 = vv[0]*vv[0] + vv[1]*vv[1] + vv[2]*vv[2] + vv[3]*vv[3];
  __shared__ float rs[256], rs2[256];
  rs[tid] = s; rs2[tid] = s2; __syncthreads();
  for (int st = 128; st > 0; st >>= 1) {
    if (tid < st) { rs[tid] += rs[tid + st]; rs2[tid] += rs2[tid + st]; }
    __syncthreads();
  }
  const float mu   = rs[0] * (1.0f / D_);
  const float var  = rs2[0] * (1.0f / D_) - mu * mu;
  const float rstd = rsqrtf(var + 1e-5f);
#pragma unroll
  for (int c = 0; c < 4; ++c) {
    int i = tid * 4 + c;
    float o = (vv[c] - mu) * rstd * g[i] + bb[i];
    Y[row * D_ + i] = o;
    if (OUTB) Yb[row * D_ + i] = f2bf(o);
  }
}

// ---------------- adaptive-LIF scan over time (seq in S, parallel B*F) ----------------
__global__ __launch_bounds__(256) void lif_scan_kernel(const float* __restrict__ ff,
                                                       const float* __restrict__ decay,
                                                       const float* __restrict__ beta,
                                                       u16* __restrict__ spikes) {
  const int idx = blockIdx.x * 256 + threadIdx.x;    // over B*F
  const int b = idx / F_, f = idx % F_;
  const float dec = decay[f], bet = beta[f];
  float vm = 0.f, a = 0.f;
  const float* pi = ff + (long long)b * S_ * F_ + f;
  u16* po = spikes + (long long)b * S_ * F_ + f;
  for (int t = 0; t < S_; ++t) {
    vm = dec * vm + pi[(long long)t * F_];
    float u  = vm - (1.0f + a);                      // THETA0 + a
    float sp = (u > 0.f) ? 1.0f : 0.0f;              // hard Heaviside (forward)
    vm *= (1.0f - sp);                               // hard reset
    a = 0.9f * a + bet * sp;                         // RHO * a + beta * s
    po[(long long)t * F_] = f2bf(sp);
  }
}

extern "C" void kernel_launch(void* const* d_in, const int* in_sizes, int n_in,
                              void* d_out, int out_size, void* d_ws, size_t ws_size,
                              hipStream_t stream) {
  (void)in_sizes; (void)n_in; (void)out_size; (void)ws_size;
  const float* x    = (const float*)d_in[0];
  const float* Wq   = (const float*)d_in[1];  const float* bq   = (const float*)d_in[2];
  const float* Wk   = (const float*)d_in[3];  const float* bk   = (const float*)d_in[4];
  const float* Wv   = (const float*)d_in[5];  const float* bv   = (const float*)d_in[6];
  const float* Wo   = (const float*)d_in[7];  const float* bo   = (const float*)d_in[8];
  const float* g1   = (const float*)d_in[9];  const float* b1   = (const float*)d_in[10];
  const float* W1   = (const float*)d_in[11]; const float* bf1  = (const float*)d_in[12];
  const float* W2   = (const float*)d_in[13]; const float* bf2  = (const float*)d_in[14];
  const float* g2   = (const float*)d_in[15]; const float* b2   = (const float*)d_in[16];
  const float* decay= (const float*)d_in[17]; const float* beta = (const float*)d_in[18];

  char* ws = (char*)d_ws;
  size_t off = 0;
  auto alloc = [&](size_t bytes) -> void* {
    void* p = ws + off; off = (off + bytes + 255) & ~(size_t)255; return p;
  };
  u16*  WqT   = (u16*)alloc((size_t)D_ * D_ * 2);
  u16*  WkT   = (u16*)alloc((size_t)D_ * D_ * 2);
  u16*  WvT   = (u16*)alloc((size_t)D_ * D_ * 2);
  u16*  WoT   = (u16*)alloc((size_t)D_ * D_ * 2);
  u16*  W1T   = (u16*)alloc((size_t)D_ * F_ * 2);   // [F, D]
  u16*  W2T   = (u16*)alloc((size_t)F_ * D_ * 2);   // [D, F]
  u16*  xb    = (u16*)alloc((size_t)B_ * S_ * D_ * 2);
  u16*  qb    = (u16*)alloc((size_t)B_ * S_ * D_ * 2);
  u16*  kb    = (u16*)alloc((size_t)B_ * S_ * D_ * 2);
  u16*  vb    = (u16*)alloc((size_t)B_ * S_ * D_ * 2);
  u16*  attnb = (u16*)alloc((size_t)B_ * S_ * D_ * 2);
  float* hbuf = (float*)alloc((size_t)B_ * S_ * D_ * 4);
  u16*  hb    = (u16*)alloc((size_t)B_ * S_ * D_ * 2);
  float* tmp  = (float*)alloc((size_t)B_ * S_ * D_ * 4);   // h_pre, then h+ffout
  float* ff   = (float*)alloc((size_t)B_ * S_ * F_ * 4);
  u16*  spk   = (u16*)alloc((size_t)B_ * S_ * F_ * 2);
  float* sc   = (float*)alloc((size_t)H_ * S_ * S_ * 4);   // per-batch scores
  u16*  pb    = (u16*)alloc((size_t)H_ * S_ * S_ * 2);     // per-batch probs (bf16)

  const int M = B_ * S_;

  // 1) weight convert+transpose, activation convert
  cvtT_bf16_kernel<<<(D_ * D_ + 255) / 256, 256, 0, stream>>>(Wq, WqT, D_, D_);
  cvtT_bf16_kernel<<<(D_ * D_ + 255) / 256, 256, 0, stream>>>(Wk, WkT, D_, D_);
  cvtT_bf16_kernel<<<(D_ * D_ + 255) / 256, 256, 0, stream>>>(Wv, WvT, D_, D_);
  cvtT_bf16_kernel<<<(D_ * D_ + 255) / 256, 256, 0, stream>>>(Wo, WoT, D_, D_);
  cvtT_bf16_kernel<<<(D_ * F_ + 255) / 256, 256, 0, stream>>>(W1, W1T, D_, F_);
  cvtT_bf16_kernel<<<(F_ * D_ + 255) / 256, 256, 0, stream>>>(W2, W2T, F_, D_);
  cvt_bf16_kernel<<<(M * D_ + 255) / 256, 256, 0, stream>>>(x, xb, M * D_);

  // 2) QKV projections (bf16 out for attention)
  gemm_bf16_wmma<128,128,2,4,false,true,false,true,false><<<dim3(D_/128, M/128, 1), 256, 0, stream>>>(
      xb, 0, D_, WqT, 0, D_, bq, nullptr, 0, 0, qb, 0, D_, D_);
  gemm_bf16_wmma<128,128,2,4,false,true,false,true,false><<<dim3(D_/128, M/128, 1), 256, 0, stream>>>(
      xb, 0, D_, WkT, 0, D_, bk, nullptr, 0, 0, kb, 0, D_, D_);
  gemm_bf16_wmma<128,128,2,4,false,true,false,true,false><<<dim3(D_/128, M/128, 1), 256, 0, stream>>>(
      xb, 0, D_, WvT, 0, D_, bv, nullptr, 0, 0, vb, 0, D_, D_);

  // 3) attention per batch (heads batched over grid.z)
  for (int b = 0; b < B_; ++b) {
    const u16* qbb = qb + (size_t)b * S_ * D_;
    const u16* kbb = kb + (size_t)b * S_ * D_;
    const u16* vbb = vb + (size_t)b * S_ * D_;
    // scores = q @ k^T (Bt = k, [S, DH] with ldb=D); skip blocks above diagonal
    gemm_bf16_wmma<128,128,2,4,false,false,false,false,true><<<dim3(S_/128, S_/128, H_), 256, 0, stream>>>(
        qbb, DH_, D_, kbb, DH_, D_, nullptr, nullptr, 0, 0, sc, (long long)S_ * S_, S_, DH_);
    softmax_causal_kernel<<<dim3(S_, H_), 256, 0, stream>>>(sc, pb);
    // attn = p @ v  (B row-major [S, DH] with ldb=D -> transpose in LDS)
    gemm_bf16_wmma<128,64,4,2,true,false,false,true,false><<<dim3(1, S_/128, H_), 256, 0, stream>>>(
        pb, (long long)S_ * S_, S_, vbb, DH_, D_, nullptr, nullptr, 0, 0,
        attnb + (size_t)b * S_ * D_, DH_, D_, S_);
  }

  // 4) output projection + residual x -> tmp; LN1 -> h (fp32) + hb (bf16)
  gemm_bf16_wmma<128,128,2,4,false,true,true,false,false><<<dim3(D_/128, M/128, 1), 256, 0, stream>>>(
      attnb, 0, D_, WoT, 0, D_, bo, x, 0, D_, tmp, 0, D_, D_);
  layernorm_kernel<true><<<M, 256, 0, stream>>>(tmp, g1, b1, hbuf, hb);

  // 5) FFN up, LIF scan, FFN down (+h residual), LN2 -> out
  gemm_bf16_wmma<128,128,2,4,false,true,false,false,false><<<dim3(F_/128, M/128, 1), 256, 0, stream>>>(
      hb, 0, D_, W1T, 0, D_, bf1, nullptr, 0, 0, ff, 0, F_, D_);
  lif_scan_kernel<<<(B_ * F_) / 256, 256, 0, stream>>>(ff, decay, beta, spk);
  gemm_bf16_wmma<128,128,2,4,false,true,true,false,false><<<dim3(D_/128, M/128, 1), 256, 0, stream>>>(
      spk, 0, F_, W2T, 0, F_, bf2, hbuf, 0, D_, tmp, 0, D_, F_);
  layernorm_kernel<false><<<M, 256, 0, stream>>>(tmp, g2, b2, (float*)d_out, nullptr);
}
